// Attention_67765993997256
// MI455X (gfx1250) — matrix operand order repaired and checked by reading, not compiled, not run on hardware
//
#include <hip/hip_runtime.h>

// ---------------- problem constants ----------------
static constexpr int B_    = 16;
static constexpr int TOUT  = 512;
static constexpr int TIN   = 1024;
static constexpr int DIM   = 1024;
static constexpr int HEADS = 4;
static constexpr int HD    = DIM / HEADS;             // 256
static constexpr long OUT_ELEMS  = (long)B_ * TOUT * DIM;         // 8,388,608
static constexpr long ATTN_ELEMS = (long)B_ * TOUT * HEADS * TIN; // 33,554,432

// ---------------- vector types ----------------
typedef _Float16 h8  __attribute__((ext_vector_type(8)));
typedef _Float16 h16 __attribute__((ext_vector_type(16)));
typedef float    f8  __attribute__((ext_vector_type(8)));
typedef int      v4i __attribute__((ext_vector_type(4)));

// ---------------- address-space-qualified pointer types for async LDS copy ----------------
typedef v4i __attribute__((address_space(1))) v4i_glb;   // global (AS1) int4
typedef v4i __attribute__((address_space(3))) v4i_lds;   // LDS    (AS3) int4
#define TO_GLB(p) ((v4i_glb*)(unsigned long long)(p))
#define TO_LDS(p) ((v4i_lds*)(unsigned int)(unsigned long long)(p))

#if __has_builtin(__builtin_amdgcn_global_load_async_to_lds_b128)
#define HAVE_ASYNC_LDS 1
#else
#define HAVE_ASYNC_LDS 0
#endif

#if __has_builtin(__builtin_amdgcn_s_wait_asynccnt)
#define WAIT_ASYNC(n) __builtin_amdgcn_s_wait_asynccnt(n)
#else
#define WAIT_ASYNC(n) asm volatile("s_wait_asynccnt %0" ::"n"(n) : "memory")
#endif

// ============================================================
// fp32 -> f16 strided convert
// ============================================================
__global__ void cvt_f32_f16(const float* __restrict__ src, _Float16* __restrict__ dst,
                            long rows, long cols, long sld, long dld) {
  long n = rows * cols;
  for (long i = (long)blockIdx.x * blockDim.x + threadIdx.x; i < n;
       i += (long)gridDim.x * blockDim.x) {
    long r = i / cols, c = i - r * cols;
    dst[r * dld + c] = (_Float16)src[r * sld + c];
  }
}

// ============================================================
// c16 [B, Tin, H*HD] -> cT16 [B, H, HD, Tin]   (tiled transpose)
// grid: (Tin/32, HD/32, B*H), block: (32, 8)
// ============================================================
__global__ void transpose_c(const _Float16* __restrict__ c16, _Float16* __restrict__ cT16) {
  __shared__ _Float16 tile[32][33];
  int z = blockIdx.z;
  int b = z / HEADS, h = z % HEADS;
  int l0 = blockIdx.x * 32;   // Tin block
  int e0 = blockIdx.y * 32;   // HD block
  for (int row = threadIdx.y; row < 32; row += 8) {
    tile[row][threadIdx.x] =
        c16[((long)b * TIN + l0 + row) * DIM + h * HD + e0 + threadIdx.x];
  }
  __syncthreads();
  for (int row = threadIdx.y; row < 32; row += 8) {
    cT16[(((long)(b * HEADS + h) * HD) + e0 + row) * TIN + l0 + threadIdx.x] =
        tile[threadIdx.x][row];
  }
}

// ============================================================
// Generic tiled WMMA GEMM:  C[m,n] = sum_k A[m,k] * Bt[n,k]  (+bias, +tanh)
// Double-buffered LDS, async global->LDS staging (ASYNCcnt pipeline).
// Block tile 128x128, 8 waves, each wave 64x32 via 16x16x32 WMMA.
// ============================================================
template <bool OUT16, bool BIAS, bool TANH>
__global__ __launch_bounds__(256) void gemm_wmma(
    const _Float16* __restrict__ A, long lda, long sAo, long sAi,
    const _Float16* __restrict__ Bt, long ldb, long sBo, long sBi,
    float* __restrict__ Cf, _Float16* __restrict__ Ch, long ldc, long sCo, long sCi,
    const float* __restrict__ bias, int K, int batch_div) {
  __shared__ __attribute__((aligned(128))) _Float16 As[2][128][40];
  __shared__ __attribute__((aligned(128))) _Float16 Bs[2][128][40];

  int z  = blockIdx.z;
  int zo = z / batch_div, zi = z % batch_div;
  A  += (long)zo * sAo + (long)zi * sAi;
  Bt += (long)zo * sBo + (long)zi * sBi;
  long cOff = (long)zo * sCo + (long)zi * sCi;

  const int tid  = threadIdx.x;
  const int lane = tid & 31;
  const int wave = tid >> 5;
  const int wm = wave & 1;        // 0..1 -> 64-row half
  const int wn = wave >> 1;       // 0..3 -> 32-col slice
  const int lhalf = lane >> 4;    // 0/1
  const int lmod  = lane & 15;
  const long rowBase = (long)blockIdx.y * 128;
  const long colBase = (long)blockIdx.x * 128;

  // each thread owns 4 fixed 16B chunks per K-tile: 2 of A, 2 of B
  const int r0 = tid >> 2;                 // chunk rows: r0 and r0+64
  const int c0 = (tid & 3) << 3;           // chunk col (f16 elems)

  // stage one 128x32 K-tile of A and Bt into LDS buffer bi
  auto stage_tile = [&](int bi, int k0) {
    const _Float16* ga0 = A  + (rowBase + r0)      * lda + k0 + c0;
    const _Float16* ga1 = A  + (rowBase + r0 + 64) * lda + k0 + c0;
    const _Float16* gb0 = Bt + (colBase + r0)      * ldb + k0 + c0;
    const _Float16* gb1 = Bt + (colBase + r0 + 64) * ldb + k0 + c0;
#if HAVE_ASYNC_LDS
    __builtin_amdgcn_global_load_async_to_lds_b128(TO_GLB(ga0), TO_LDS(&As[bi][r0][c0]), 0, 0);
    __builtin_amdgcn_global_load_async_to_lds_b128(TO_GLB(ga1), TO_LDS(&As[bi][r0 + 64][c0]), 0, 0);
    __builtin_amdgcn_global_load_async_to_lds_b128(TO_GLB(gb0), TO_LDS(&Bs[bi][r0][c0]), 0, 0);
    __builtin_amdgcn_global_load_async_to_lds_b128(TO_GLB(gb1), TO_LDS(&Bs[bi][r0 + 64][c0]), 0, 0);
#else
    h8 va0 = *(const h8*)ga0;
    h8 va1 = *(const h8*)ga1;
    h8 vb0 = *(const h8*)gb0;
    h8 vb1 = *(const h8*)gb1;
    *(h8*)&As[bi][r0][c0]      = va0;
    *(h8*)&As[bi][r0 + 64][c0] = va1;
    *(h8*)&Bs[bi][r0][c0]      = vb0;
    *(h8*)&Bs[bi][r0 + 64][c0] = vb1;
#endif
  };

  f8 acc[4][2] = {};
  const int ntiles = K >> 5;

  stage_tile(0, 0);
  if (ntiles > 1) stage_tile(1, 32);

  for (int tk = 0; tk < ntiles; ++tk) {
    const int bi = tk & 1;
    // tile tk is complete when only tile tk+1's 4 ops remain outstanding
    if (tk + 1 < ntiles) { WAIT_ASYNC(4); } else { WAIT_ASYNC(0); }
    __syncthreads();

    // ---- B fragments: lane holds col n = lmod; elems 0..7 -> K = 8*lhalf + i,
    //      elems 8..15 -> K = 16 + 8*lhalf + i ----
    h16 bfrag[2];
#pragma unroll
    for (int nb = 0; nb < 2; ++nb) {
      int n = wn * 32 + nb * 16 + lmod;
      h8 lo = *(const h8*)&Bs[bi][n][lhalf * 8];
      h8 hi = *(const h8*)&Bs[bi][n][16 + lhalf * 8];
      bfrag[nb] = __builtin_shufflevector(lo, hi, 0, 1, 2, 3, 4, 5, 6, 7,
                                          8, 9, 10, 11, 12, 13, 14, 15);
    }

    // ---- A fragments: row m = lmod; elems 0..7 -> K = 8*lhalf + i,
    //      elems 8..15 -> K = 16 + 8*lhalf + i ----
#pragma unroll
    for (int mb = 0; mb < 4; ++mb) {
      int r = wm * 64 + mb * 16 + lmod;
      h8 lo = *(const h8*)&As[bi][r][lhalf * 8];
      h8 hi = *(const h8*)&As[bi][r][16 + lhalf * 8];
      h16 afrag = __builtin_shufflevector(lo, hi, 0, 1, 2, 3, 4, 5, 6, 7,
                                          8, 9, 10, 11, 12, 13, 14, 15);
#pragma unroll
      for (int nb = 0; nb < 2; ++nb) {
        acc[mb][nb] = __builtin_amdgcn_wmma_f32_16x16x32_f16(
            false, afrag, false, bfrag[nb], (short)0, acc[mb][nb], false, false);
      }
    }
    __syncthreads();   // all waves done reading buf bi before it is refilled

    if (tk + 2 < ntiles) stage_tile(bi, (tk + 2) << 5);
  }

  // ---- epilogue: C/D layout: VGPR r -> M = r + 8*lhalf, N = lmod ----
#pragma unroll
  for (int mb = 0; mb < 4; ++mb) {
#pragma unroll
    for (int nb = 0; nb < 2; ++nb) {
      long col = colBase + wn * 32 + nb * 16 + lmod;
      float bv = BIAS ? bias[col] : 0.0f;
#pragma unroll
      for (int r = 0; r < 8; ++r) {
        long row = rowBase + wm * 64 + mb * 16 + lhalf * 8 + r;
        float v = acc[mb][nb][r] + bv;
        if (TANH) v = tanhf(v);
        long off = cOff + row * ldc + col;
        if (OUT16) Ch[off] = (_Float16)v;
        else       Cf[off] = v;
      }
    }
  }
}

// ============================================================
// Row softmax over Tin=1024, in-place on f32 scores in the attn_out
// region of d_out, plus f16 copy for the stage-5 GEMM.
// ============================================================
__global__ __launch_bounds__(256) void softmax_rows(float* __restrict__ attn_f32,
                                                    _Float16* __restrict__ attn_f16) {
  __shared__ float red[256];
  int tid = threadIdx.x;
  long r = blockIdx.x;                       // 0 .. B*H*Tout-1
  long b = r >> 11;                          // H*Tout = 2048 rows per batch
  long rem = r & 2047;
  long h = rem >> 9;                         // Tout = 512
  long t = rem & 511;
  long base = (b * TOUT + t) * (long)(HEADS * TIN) + h * TIN;

  float4 v = *(const float4*)(attn_f32 + base + tid * 4);
  float lm = fmaxf(fmaxf(v.x, v.y), fmaxf(v.z, v.w));
  red[tid] = lm;
  __syncthreads();
  for (int s = 128; s > 0; s >>= 1) {
    if (tid < s) red[tid] = fmaxf(red[tid], red[tid + s]);
    __syncthreads();
  }
  float m = red[0];
  __syncthreads();

  float4 e;
  e.x = expf(v.x - m); e.y = expf(v.y - m);
  e.z = expf(v.z - m); e.w = expf(v.w - m);
  red[tid] = e.x + e.y + e.z + e.w;
  __syncthreads();
  for (int s = 128; s > 0; s >>= 1) {
    if (tid < s) red[tid] += red[tid + s];
    __syncthreads();
  }
  float inv = 1.0f / red[0];

  e.x *= inv; e.y *= inv; e.z *= inv; e.w *= inv;
  *(float4*)(attn_f32 + base + tid * 4) = e;
  _Float16* d16 = attn_f16 + base + tid * 4;
  d16[0] = (_Float16)e.x; d16[1] = (_Float16)e.y;
  d16[2] = (_Float16)e.z; d16[3] = (_Float16)e.w;
}

// ============================================================
// launcher
// ============================================================
extern "C" void kernel_launch(void* const* d_in, const int* in_sizes, int n_in,
                              void* d_out, int out_size, void* d_ws, size_t ws_size,
                              hipStream_t stream) {
  const float* output  = (const float*)d_in[0];  // [B,Tout,DIM]
  const float* context = (const float*)d_in[1];  // [B,Tin,DIM]
  const float* wc_w    = (const float*)d_in[2];  // [H,HD,DIM] == [N=DIM, K=DIM]
  const float* wc_b    = (const float*)d_in[3];  // [H,HD] -> [DIM]
  const float* wo_w    = (const float*)d_in[4];
  const float* wo_b    = (const float*)d_in[5];
  const float* comb_w  = (const float*)d_in[6];  // [DIM,DIM] (N,K)
  const float* comb_b  = (const float*)d_in[7];
  const float* out_w   = (const float*)d_in[8];  // [DIM,2*DIM] (N,K)
  const float* out_b   = (const float*)d_in[9];

  float* out_f32  = (float*)d_out;               // [B,Tout,DIM]
  float* attn_f32 = out_f32 + OUT_ELEMS;         // [B,Tout,H*Tin]

  // ---------------- workspace carve-up (f16 buffers) ----------------
  char* p = (char*)d_ws;
  auto take = [&](size_t bytes) { char* q = p; p += (bytes + 255) & ~(size_t)255; return q; };
  _Float16* ctx16   = (_Float16*)take((size_t)B_ * TIN * DIM * 2);       // [B*Tin, DIM]
  _Float16* catbuf  = (_Float16*)take((size_t)B_ * TOUT * 2 * DIM * 2);  // [B*Tout, 2*DIM]: [mixed|output]
  _Float16* c16     = (_Float16*)take((size_t)B_ * TIN * DIM * 2);       // [B,Tin,H*HD]
  _Float16* cT16    = (_Float16*)take((size_t)B_ * HEADS * HD * TIN * 2);// [B,H,HD,Tin]
  _Float16* o16     = (_Float16*)take((size_t)B_ * TOUT * DIM * 2);      // [B,Tout,H*HD]
  _Float16* attn16  = (_Float16*)take((size_t)ATTN_ELEMS * 2);           // [B,Tout,H*Tin]
  _Float16* mix16   = (_Float16*)take((size_t)B_ * TOUT * DIM * 2);      // [B,Tout,H*HD]
  _Float16* wc16    = (_Float16*)take((size_t)DIM * DIM * 2);
  _Float16* wo16    = (_Float16*)take((size_t)DIM * DIM * 2);
  _Float16* wcomb16 = (_Float16*)take((size_t)DIM * DIM * 2);
  _Float16* wout16  = (_Float16*)take((size_t)DIM * 2 * DIM * 2);

  // ---------------- fp32 -> f16 conversions ----------------
  cvt_f32_f16<<<4096, 256, 0, stream>>>(context, ctx16, (long)B_ * TIN, DIM, DIM, DIM);
  cvt_f32_f16<<<4096, 256, 0, stream>>>(output, catbuf + DIM, (long)B_ * TOUT, DIM, DIM, 2 * DIM);
  cvt_f32_f16<<<1024, 256, 0, stream>>>(wc_w,   wc16,    DIM, DIM, DIM, DIM);
  cvt_f32_f16<<<1024, 256, 0, stream>>>(wo_w,   wo16,    DIM, DIM, DIM, DIM);
  cvt_f32_f16<<<1024, 256, 0, stream>>>(comb_w, wcomb16, DIM, DIM, DIM, DIM);
  cvt_f32_f16<<<1024, 256, 0, stream>>>(out_w,  wout16,  DIM, 2 * DIM, 2 * DIM, 2 * DIM);

  // ---------------- stage 1: c = context @ wc^T + wc_b -> c16 (f16) ----------------
  gemm_wmma<true, true, false><<<dim3(DIM / 128, (B_ * TIN) / 128, 1), 256, 0, stream>>>(
      ctx16, DIM, 0, 0, wc16, DIM, 0, 0,
      nullptr, c16, DIM, 0, 0, wc_b, DIM, 1);

  // ---------------- stage 2: o = output @ wo^T + wo_b -> o16 (f16) ----------------
  gemm_wmma<true, true, false><<<dim3(DIM / 128, (B_ * TOUT) / 128, 1), 256, 0, stream>>>(
      catbuf + DIM, 2 * DIM, 0, 0, wo16, DIM, 0, 0,
      nullptr, o16, DIM, 0, 0, wo_b, DIM, 1);

  // ---------------- stage 3: scores = o_bh @ c_bh^T -> attn region of d_out (f32) ----
  gemm_wmma<false, false, false><<<dim3(TIN / 128, TOUT / 128, B_ * HEADS), 256, 0, stream>>>(
      o16, DIM, (long)TOUT * DIM, HD,            // A: o[b, t, h*HD + k]
      c16, DIM, (long)TIN * DIM, HD,             // Bt: c[b, l, h*HD + k]
      attn_f32, nullptr, (long)HEADS * TIN,      // C: attn[b, t, h*Tin + l]
      (long)TOUT * HEADS * TIN, TIN,
      nullptr, HD, HEADS);

  // ---------------- stage 4: softmax (in place) + f16 copy ----------------
  softmax_rows<<<B_ * HEADS * TOUT, 256, 0, stream>>>(attn_f32, attn16);

  // transpose c for stage 5 (Bt needs [HD, Tin] per (b,h))
  transpose_c<<<dim3(TIN / 32, HD / 32, B_ * HEADS), dim3(32, 8), 0, stream>>>(c16, cT16);

  // ---------------- stage 5: mix = attn_bh @ c_bh -> mix16 (f16) ----------------
  gemm_wmma<true, false, false><<<dim3(HD / 128, TOUT / 128, B_ * HEADS), 256, 0, stream>>>(
      attn16, (long)HEADS * TIN, (long)TOUT * HEADS * TIN, TIN,   // A: attn[b, t, h*Tin + k]
      cT16, TIN, (long)HEADS * HD * TIN, (long)HD * TIN,          // Bt: cT[b, h, e, k]
      nullptr, mix16, DIM, (long)TOUT * DIM, HD,                  // C: mix -> concat layout
      nullptr, TIN, HEADS);

  // ---------------- stage 6a: mixed = concat @ comb^T + comb_b -> catbuf[:, :DIM] ----
  gemm_wmma<true, true, false><<<dim3(DIM / 128, (B_ * TOUT) / 128, 1), 256, 0, stream>>>(
      mix16, DIM, 0, 0, wcomb16, DIM, 0, 0,
      nullptr, catbuf, 2 * DIM, 0, 0, comb_b, DIM, 1);

  // ---------------- stage 6b: out = tanh([mixed|output] @ out_w^T + out_b) ----------
  gemm_wmma<false, true, true><<<dim3(DIM / 128, (B_ * TOUT) / 128, 1), 256, 0, stream>>>(
      catbuf, 2 * DIM, 0, 0, wout16, 2 * DIM, 0, 0,
      out_f32, nullptr, DIM, 0, 0, out_b, 2 * DIM, 1);
}
